// GATModel_6124623364709
// MI455X (gfx1250) — matrix operand order repaired and checked by reading, compiled, not run on hardware
//
#include <hip/hip_runtime.h>
#include <hip/hip_bf16.h>
#include <cstdint>
#include <cstddef>

#define N_NODES 50000
#define N_EDGES 800000
#define C_IN    128
#define C_HID   256
#define SLOPE   0.2f

typedef __attribute__((ext_vector_type(16))) __bf16 v16bf;
typedef __attribute__((ext_vector_type(8)))  float  v8f;

// ---------- order-preserving float<->uint encoding for atomicMax on signed floats ----------
__device__ __forceinline__ unsigned enc_f32(float x) {
  unsigned u = __float_as_uint(x);
  return (u & 0x80000000u) ? ~u : (u | 0x80000000u);
}
__device__ __forceinline__ float dec_f32(unsigned u) {
  return __uint_as_float((u & 0x80000000u) ? (u & 0x7FFFFFFFu) : ~u);
}

// =========================================================================================
// One-time per layer: Wt[n*Cin + k] = bf16(W[k*Cout + n])  for both Wl and Wr.
// Makes each WMMA B-fragment a single contiguous 32-byte load in the GEMM hot loop.
// =========================================================================================
__global__ void w_to_bf16t(const float* __restrict__ Wl, const float* __restrict__ Wr,
                           __bf16* __restrict__ Wlt, __bf16* __restrict__ Wrt,
                           int Cin, int Cout)
{
  int idx = blockIdx.x * 256 + threadIdx.x;          // idx over Cin*Cout, contiguous in k
  if (idx >= Cin * Cout) return;
  int n = idx / Cin, k = idx - n * Cin;
  Wlt[idx] = (__bf16)Wl[(size_t)k * Cout + n];
  Wrt[idx] = (__bf16)Wr[(size_t)k * Cout + n];
}

// =========================================================================================
// Fused dual GEMM:  Yl = X @ Wl,  Yr = X @ Wr   (fp32 X/Y, bf16 WMMA compute).
// W is pre-transposed bf16 (column-major: Wt[n][k]).
//
// Block = 256 threads = 8 waves. Block tile: 64 rows x 128 cols.
//   - stage 64 x Cin of X into LDS once (bf16, padded stride)
//   - wave w owns columns [128*bx + 16w, +16)
//   - per K-step(32): B frags = 2 x v16bf contiguous loads; 4 M-subtiles -> 8 WMMAs
// gridDim = (Cout/128, ceil(n_rows/64)); n_rows % 16 == 0 -> uniform subtile guards.
// =========================================================================================
__global__ __launch_bounds__(256)
void gemm_dual_bf16(const float* __restrict__ X,
                    const __bf16* __restrict__ Wlt, const __bf16* __restrict__ Wrt,
                    float* __restrict__ Yl, float* __restrict__ Yr,
                    int n_rows, int Cin, int Cout)
{
  __shared__ __bf16 As[64 * (C_HID + 8)];   // 64 x 264 bf16 = 33 KB
  const int LDA  = Cin + 8;
  const int row0 = blockIdx.y * 64;
  const int tid  = threadIdx.x;

  // ---- stage 64 x Cin tile of X into LDS as bf16 (float4 reads, coalesced along k) ----
  const int kvec = Cin >> 2;
  for (int idx = tid; idx < 64 * kvec; idx += 256) {
    int r  = idx / kvec;
    int k4 = (idx - r * kvec) << 2;
    float4 v = make_float4(0.f, 0.f, 0.f, 0.f);
    if (row0 + r < n_rows)
      v = *(const float4*)&X[(size_t)(row0 + r) * Cin + k4];
    __bf16* p = &As[r * LDA + k4];
    p[0] = (__bf16)v.x; p[1] = (__bf16)v.y; p[2] = (__bf16)v.z; p[3] = (__bf16)v.w;
  }
  __syncthreads();

  const int wave = tid >> 5;
  const int lane = tid & 31;
  const int l    = lane & 15;   // A-row / B,D-column selector within half-wave
  const int hi   = lane >> 4;   // half-wave selector
  const int n    = blockIdx.x * 128 + wave * 16 + l;   // output column

  const __bf16* Wlp = Wlt + (size_t)n * Cin;           // column n, contiguous in k
  const __bf16* Wrp = Wrt + (size_t)n * Cin;

  v8f accl[4] = {};
  v8f accr[4] = {};

  for (int k0 = 0; k0 < Cin; k0 += 32) {
    if (k0 + 32 < Cin) {        // keep the next W panels warm
      __builtin_prefetch(&Wlp[k0 + 32], 0, 1);
      __builtin_prefetch(&Wrp[k0 + 32], 0, 1);
    }

    // B 32x16 bf16: lanes 0-15 K=k0..k0+15, lanes 16-31 K=k0+16..k0+31 -> contiguous 32B
    v16bf bl = *(const v16bf*)&Wlp[k0 + hi * 16];
    v16bf br = *(const v16bf*)&Wrp[k0 + hi * 16];

    // ---- 4 M-subtiles reuse both B fragments ----
    #pragma unroll
    for (int mt = 0; mt < 4; ++mt) {
      const __bf16* Ap = &As[(mt * 16 + l) * LDA];
      v16bf a;
      // A 16x32 bf16: lanes 0-15 K=[0..15]; lanes 16-31 K=[16..31]
      #pragma unroll
      for (int i = 0; i < 4; ++i) {
        int k = k0 + hi * 16 + 2 * i;
        a[2 * i]     = Ap[k];
        a[2 * i + 1] = Ap[k + 1];
      }
      #pragma unroll
      for (int i = 4; i < 8; ++i) {
        int k = k0 + hi * 16 + 8 + 2 * (i - 4);
        a[2 * i]     = Ap[k];
        a[2 * i + 1] = Ap[k + 1];
      }
      accl[mt] = __builtin_amdgcn_wmma_f32_16x16x32_bf16(false, a, false, bl,
                                                         (short)0, accl[mt], false, false);
      accr[mt] = __builtin_amdgcn_wmma_f32_16x16x32_bf16(false, a, false, br,
                                                         (short)0, accr[mt], false, false);
    }
  }

  // ---- store: D 16x16 f32: VGPR r -> M = r + 8*hi, N = l; uniform per-subtile guard ----
  #pragma unroll
  for (int mt = 0; mt < 4; ++mt) {
    int mbase = row0 + mt * 16;
    if (mbase < n_rows) {
      #pragma unroll
      for (int r = 0; r < 8; ++r) {
        size_t off = (size_t)(mbase + r + hi * 8) * Cout + n;
        Yl[off] = accl[mt][r];
        Yr[off] = accr[mt][r];
      }
    }
  }
}

// ============================ per-layer init ============================
__global__ void layer_init(float* __restrict__ agg, unsigned* __restrict__ emax,
                           float* __restrict__ denom, int total)
{
  int i = blockIdx.x * 256 + threadIdx.x;
  if (i < total)   agg[i]   = 0.0f;
  if (i < N_NODES) { emax[i] = 0u; denom[i] = 0.0f; }   // enc==0 is below every real float
}

// ======== e_k = att . leaky_relu(xl[src] + xr[dst] + ea*We); segment-max via atomics ========
__global__ __launch_bounds__(256)
void edge_score(const long long* __restrict__ ei, const float* __restrict__ ea,
                const float* __restrict__ xl, const float* __restrict__ xr,
                const float* __restrict__ We, const float* __restrict__ att,
                float* __restrict__ e_out, unsigned* __restrict__ emax_enc, int C)
{
  int edge = blockIdx.x * 8 + (threadIdx.x >> 5);       // one wave32 per edge
  if (edge >= N_EDGES) return;
  int lane = threadIdx.x & 31;
  long long s = ei[edge];
  long long d = ei[N_EDGES + edge];
  float eav = ea[edge];
  const float* pl = xl + (size_t)s * C;
  const float* pr = xr + (size_t)d * C;
  float acc = 0.0f;
  for (int j = 0; j < C; j += 128) {                    // per-lane float4, wave-coalesced
    int c = j + lane * 4;
    float4 vl = *(const float4*)&pl[c];
    float4 vr = *(const float4*)&pr[c];
    float4 vw = *(const float4*)&We[c];
    float4 va = *(const float4*)&att[c];
    float m0 = vl.x + vr.x + eav * vw.x; m0 = (m0 > 0.f) ? m0 : m0 * SLOPE;
    float m1 = vl.y + vr.y + eav * vw.y; m1 = (m1 > 0.f) ? m1 : m1 * SLOPE;
    float m2 = vl.z + vr.z + eav * vw.z; m2 = (m2 > 0.f) ? m2 : m2 * SLOPE;
    float m3 = vl.w + vr.w + eav * vw.w; m3 = (m3 > 0.f) ? m3 : m3 * SLOPE;
    acc += m0 * va.x + m1 * va.y + m2 * va.z + m3 * va.w;
  }
  #pragma unroll
  for (int off = 16; off > 0; off >>= 1) acc += __shfl_xor(acc, off, 32);
  if (lane == 0) {
    e_out[edge] = acc;
    atomicMax(&emax_enc[d], enc_f32(acc));
  }
}

// ================= ex = exp(e - emax[dst]); segment-sum denominator =================
__global__ void edge_exp(const long long* __restrict__ ei, float* __restrict__ e,
                         const unsigned* __restrict__ emax_enc, float* __restrict__ denom)
{
  int k = blockIdx.x * 256 + threadIdx.x;
  if (k >= N_EDGES) return;
  long long d = ei[N_EDGES + k];
  float ex = __expf(e[k] - dec_f32(emax_enc[d]));
  e[k] = ex;
  atomicAdd(&denom[d], ex);
}

// ================= agg[dst] += (ex/denom[dst]) * xl[src] =================
__global__ __launch_bounds__(256)
void edge_aggregate(const long long* __restrict__ ei, const float* __restrict__ e,
                    const float* __restrict__ denom, const float* __restrict__ xl,
                    float* __restrict__ agg, int C)
{
  int edge = blockIdx.x * 8 + (threadIdx.x >> 5);       // one wave32 per edge
  if (edge >= N_EDGES) return;
  int lane = threadIdx.x & 31;
  long long s = ei[edge];
  long long d = ei[N_EDGES + edge];
  float alpha = e[edge] / denom[d];
  const float* pl = xl + (size_t)s * C;
  float*       pd = agg + (size_t)d * C;
  for (int j = 0; j < C; j += 128) {
    int c = j + lane * 4;
    float4 vl = *(const float4*)&pl[c];
    atomicAdd(&pd[c + 0], alpha * vl.x);
    atomicAdd(&pd[c + 1], alpha * vl.y);
    atomicAdd(&pd[c + 2], alpha * vl.z);
    atomicAdd(&pd[c + 3], alpha * vl.w);
  }
}

// ================= x = relu(agg + bias), float4 =================
__global__ void finalize_relu_bias(const float* __restrict__ agg, const float* __restrict__ bias,
                                   float* __restrict__ xout, int total4)
{
  int i4 = blockIdx.x * 256 + threadIdx.x;
  if (i4 >= total4) return;
  int base = i4 * 4;
  int c = base & (C_HID - 1);
  float4 a = *(const float4*)&agg[base];
  float4 b = *(const float4*)&bias[c];
  float4 v;
  v.x = fmaxf(a.x + b.x, 0.f);
  v.y = fmaxf(a.y + b.y, 0.f);
  v.z = fmaxf(a.z + b.z, 0.f);
  v.w = fmaxf(a.w + b.w, 0.f);
  *(float4*)&xout[base] = v;
}

// ================= global max pool over nodes (inputs are post-ReLU, >= 0) =================
__global__ void pool_init(unsigned* __restrict__ pool)
{
  if (threadIdx.x < C_HID) pool[threadIdx.x] = 0u;
}
__global__ __launch_bounds__(256)
void pool_max(const float* __restrict__ x, unsigned* __restrict__ pool)
{
  int c  = threadIdx.x;                 // channel (coalesced across lanes)
  int r0 = blockIdx.x * 256;
  float m = 0.0f;
  for (int r = r0; r < r0 + 256 && r < N_NODES; ++r)
    m = fmaxf(m, x[(size_t)r * C_HID + c]);
  atomicMax(&pool[c], __float_as_uint(m));   // non-negative floats: uint order == float order
}

// ================= MLP head: 256 -> 10 (relu) -> 1 =================
__global__ void head_mlp(const unsigned* __restrict__ pool,
                         const float* __restrict__ Wh1, const float* __restrict__ bh1,
                         const float* __restrict__ Wh2, const float* __restrict__ bh2,
                         float* __restrict__ out)
{
  __shared__ float h[10];
  int j = threadIdx.x;
  if (j < 10) {
    float acc = bh1[j];
    for (int c = 0; c < C_HID; ++c) acc += __uint_as_float(pool[c]) * Wh1[c * 10 + j];
    h[j] = acc > 0.0f ? acc : 0.0f;
  }
  __syncthreads();
  if (j == 0) {
    float acc = bh2[0];
    for (int t = 0; t < 10; ++t) acc += h[t] * Wh2[t];
    out[0] = acc;
  }
}

// =========================================================================================
extern "C" void kernel_launch(void* const* d_in, const int* in_sizes, int n_in,
                              void* d_out, int out_size, void* d_ws, size_t ws_size,
                              hipStream_t stream)
{
  (void)in_sizes; (void)n_in; (void)out_size; (void)ws_size;

  const float*     features = (const float*)d_in[0];
  const long long* ei       = (const long long*)d_in[1];   // int64 [2, E]
  const float*     ea       = (const float*)d_in[2];

  const float* Wl[3]   = { (const float*)d_in[3],  (const float*)d_in[8],  (const float*)d_in[13] };
  const float* Wr[3]   = { (const float*)d_in[4],  (const float*)d_in[9],  (const float*)d_in[14] };
  const float* We[3]   = { (const float*)d_in[5],  (const float*)d_in[10], (const float*)d_in[15] };
  const float* att[3]  = { (const float*)d_in[6],  (const float*)d_in[11], (const float*)d_in[16] };
  const float* bias[3] = { (const float*)d_in[7],  (const float*)d_in[12], (const float*)d_in[17] };
  const float* Wh1 = (const float*)d_in[18];
  const float* bh1 = (const float*)d_in[19];
  const float* Wh2 = (const float*)d_in[20];
  const float* bh2 = (const float*)d_in[21];

  // ---------------- workspace layout (all chunks 32B-aligned) ----------------
  const size_t SZ = (size_t)N_NODES * C_HID;            // 12.8M floats per node-feature buffer
  float*    ws    = (float*)d_ws;
  float*    xl    = ws;                                 // [N,256]
  float*    xr    = xl + SZ;                            // [N,256]
  float*    agg   = xr + SZ;                            // [N,256]
  float*    xbuf  = agg + SZ;                           // [N,256] layer activations
  float*    e     = xbuf + SZ;                          // [E]
  float*    denom = e + N_EDGES;                        // [N]
  unsigned* emax  = (unsigned*)(denom + N_NODES);       // [N]
  unsigned* pool  = emax + N_NODES;                     // [256]
  __bf16*   Wlt   = (__bf16*)(pool + 256);              // [256*256] bf16, column-major
  __bf16*   Wrt   = Wlt + (size_t)C_HID * C_HID;        // [256*256] bf16, column-major

  const int total      = N_NODES * C_HID;
  const int blk_total  = (total + 255) / 256;           // 50000
  const int blk_tot4   = (total / 4 + 255) / 256;       // 12500
  const int blk_edgeW  = (N_EDGES + 7) / 8;             // 100000 (1 wave/edge)
  const int blk_edgeT  = (N_EDGES + 255) / 256;         // 3125
  const dim3 gemm_grid(C_HID / 128, (N_NODES + 63) / 64);  // (2, 782)

  for (int layer = 0; layer < 3; ++layer) {
    const float* Xin = (layer == 0) ? features : xbuf;
    const int    Cin = (layer == 0) ? C_IN : C_HID;

    w_to_bf16t<<<(Cin * C_HID + 255) / 256, 256, 0, stream>>>(Wl[layer], Wr[layer],
                                                              Wlt, Wrt, Cin, C_HID);
    gemm_dual_bf16<<<gemm_grid, 256, 0, stream>>>(Xin, Wlt, Wrt,
                                                  xl, xr, N_NODES, Cin, C_HID);

    layer_init<<<blk_total, 256, 0, stream>>>(agg, emax, denom, total);
    edge_score<<<blk_edgeW, 256, 0, stream>>>(ei, ea, xl, xr, We[layer], att[layer],
                                              e, emax, C_HID);
    edge_exp<<<blk_edgeT, 256, 0, stream>>>(ei, e, emax, denom);
    edge_aggregate<<<blk_edgeW, 256, 0, stream>>>(ei, e, denom, xl, agg, C_HID);
    finalize_relu_bias<<<blk_tot4, 256, 0, stream>>>(agg, bias[layer], xbuf, total / 4);
  }

  pool_init<<<1, 256, 0, stream>>>(pool);
  pool_max<<<(N_NODES + 255) / 256, 256, 0, stream>>>(xbuf, pool);
  head_mlp<<<1, 32, 0, stream>>>(pool, Wh1, bh1, Wh2, bh2, (float*)d_out);
}
